// get_model_23124103921971
// MI455X (gfx1250) — compile-verified, matrix operand order checked by reading, not run on hardware
//
#include <hip/hip_runtime.h>
#include <hip/hip_bf16.h>

// ---------------------------------------------------------------------------
// MI455X (gfx1250) implementation.
//  - Big GEMMs (soap MLP over 128000 rows, per-batch mask GEMM) use
//    v_wmma_f32_16x16x32_f16 (f16 in, f32 accumulate). BN after every layer
//    renormalizes, inputs are O(1), mask entries are exactly 0/1 in f16.
//  - Weight panels are staged into LDS with GLOBAL_LOAD_ASYNC_TO_LDS_B128
//    (ASYNCcnt-tracked) and consumed as ds_load_b128 B-fragments; the GEMM
//    inner loop batches the 4 B-fragment LDS loads (single wait) and
//    prefetches the next k-step A fragment behind the 4 WMMAs.
//  - Linear biases are dropped: a per-channel constant cancels exactly in the
//    batch-norm that immediately follows every linear layer (except lt4).
//  - Tiny 256-row heads (mbtr / res / lt1) are fused FC+BN+ReLU VALU kernels.
// ---------------------------------------------------------------------------

typedef __attribute__((ext_vector_type(16))) _Float16 v16h;
typedef __attribute__((ext_vector_type(8)))  _Float16 v8h;
typedef __attribute__((ext_vector_type(8)))  float    v8f;

#define DEV __device__ __forceinline__

static constexpr int   Bsz   = 256;
static constexpr int   NPTS  = 500;
static constexpr int   ROWS  = Bsz * NPTS;   // 128000
static constexpr float EPSF  = 1e-5f;

// ---------------------------------------------------------------- WMMA utils
DEV v16h mk16(v8h lo, v8h hi) {
  v16h r;
#pragma unroll
  for (int e = 0; e < 8; ++e) { r[e] = lo[e]; r[e + 8] = hi[e]; }
  return r;
}
// Load a 16x32 f16 fragment (A layout, or B from K-contiguous storage).
// p must point at (row (lane&15)) * ld + k0 + ((lane>>4)*8).
DEV v16h load_frag(const _Float16* p) {
  v8h lo = *(const v8h*)p;
  v8h hi = *(const v8h*)(p + 16);
  return mk16(lo, hi);
}
DEV v8f wmma_f16(v16h a, v16h b, v8f c) {
  return __builtin_amdgcn_wmma_f32_16x16x32_f16(false, a, false, b,
                                                (short)0, c, false, false);
}

// ------------------------------------------------------------- prep kernels
__global__ void tsys_kernel(const float* __restrict__ x, float* __restrict__ out) {
  const int b = blockIdx.x, t = threadIdx.x;
  float v = 0.f;
  for (int i = t; i < NPTS; i += 256) v += x[((size_t)b * 157 + 6) * 501 + i];
  __shared__ float red[256];
  red[t] = v; __syncthreads();
  for (int s = 128; s > 0; s >>= 1) { if (t < s) red[t] += red[t + s]; __syncthreads(); }
  if (t == 0) out[256 + b] = red[0] * (1.f / (float)NPTS);
}

__global__ void soap_cvt(const float* __restrict__ x, _Float16* __restrict__ A0) {
  const size_t N = (size_t)ROWS * 160;
  size_t idx = (size_t)blockIdx.x * 256 + threadIdx.x;
  if (idx >= N) return;
  const int    k   = (int)(idx % 160);
  const size_t row = idx / 160;
  const int    i   = (int)(row % NPTS);
  const size_t b   = row / NPTS;
  float v = 0.f;
  if (k < 147) v = x[(b * 157 + 10 + k) * 501 + i];
  A0[idx] = (_Float16)v;
}

__global__ void xyz_sq(const float* __restrict__ x, float* __restrict__ xyzP,
                       float* __restrict__ sqB) {
  const int idx = blockIdx.x * 256 + threadIdx.x;
  if (idx >= ROWS) return;
  const int i = idx % NPTS;
  const size_t b = idx / NPTS;
  const float a = x[(b * 157 + 3) * 501 + i];
  const float c = x[(b * 157 + 4) * 501 + i];
  const float d = x[(b * 157 + 5) * 501 + i];
  xyzP[(size_t)idx * 3 + 0] = a;
  xyzP[(size_t)idx * 3 + 1] = c;
  xyzP[(size_t)idx * 3 + 2] = d;
  sqB[idx] = a * a + c * c + d * d;
}

__global__ void mbtr_cvt(const float* __restrict__ x, float* __restrict__ G0) {
  const int idx = blockIdx.x * 256 + threadIdx.x;
  if (idx >= Bsz * 157) return;
  const int ch = idx % 157;
  const size_t b = idx / 157;
  G0[idx] = x[(b * 157 + ch) * 501 + 500];
}

__global__ void wcvt_f16(const float* __restrict__ Wsrc, int din, int Kpad, int dout,
                         _Float16* __restrict__ Wdst) {
  const int idx = blockIdx.x * 256 + threadIdx.x;
  if (idx >= dout * Kpad) return;
  const int k = idx % Kpad, o = idx / Kpad;
  const float v = (k < din) ? Wsrc[(size_t)o * din + k] : 0.f;
  Wdst[idx] = (_Float16)v;
}

__global__ void zero_f32(float* p, int n) {
  const int i = blockIdx.x * 256 + threadIdx.x;
  if (i < n) p[i] = 0.f;
}

// ------------------------------------------------------------- big WMMA GEMM
// Y[row, o] = sum_k A[row,k] * W[o,k].
// One block = 8 waves = 8 row-tiles (128 rows) x one 64-column weight panel.
// The panel is staged into LDS once per block via async global->LDS copies.
__global__ __launch_bounds__(256) void gemm_wmma_f16(
    const _Float16* __restrict__ A, int lda,
    const _Float16* __restrict__ Wm, int ldw,
    float* __restrict__ Y, int dout, int K) {
  __shared__ __align__(16) _Float16 ldsB[256 * 64];   // 32 KB max (K<=256)

  const int tid  = threadIdx.x;
  const int wave = tid >> 5;
  const int lane = tid & 31;
  const int col0 = blockIdx.y * 64;

  // ---- async stage of the 64-column weight panel (flat copy, K-contiguous)
  {
    const size_t bytes = (size_t)64 * K * 2;          // 20480 or 32768
    const char* gsrc = (const char*)(Wm + (size_t)col0 * ldw);
    const unsigned lbase = (unsigned)(uintptr_t)(&ldsB[0]);   // LDS byte offset
    for (size_t off = (size_t)tid * 16; off < bytes; off += 256 * 16) {
      const unsigned ldst = lbase + (unsigned)off;
      const void* gaddr = (const void*)(gsrc + off);
      asm volatile("global_load_async_to_lds_b128 %0, %1, off"
                   :: "v"(ldst), "v"(gaddr) : "memory");
    }
    asm volatile("s_wait_asynccnt 0x0" ::: "memory");
  }
  __syncthreads();

  const int row0 = (blockIdx.x * 8 + wave) * 16;
  const int l    = lane & 15;
  const int hk   = (lane >> 4) << 3;
  const _Float16* aRow = A + (size_t)(row0 + l) * lda + hk;

  v8f acc[4] = {};
  v16h af = load_frag(aRow);              // k-step 0 A fragment
  for (int k0 = 0; k0 < K; k0 += 32) {
    // Batch the 4 LDS B-fragment loads (one clause / one wait), then prefetch
    // the next k-step A fragment so its global latency hides behind 4 WMMAs.
    v16h bf0 = load_frag(&ldsB[(0 * 16 + l) * K + k0 + hk]);
    v16h bf1 = load_frag(&ldsB[(1 * 16 + l) * K + k0 + hk]);
    v16h bf2 = load_frag(&ldsB[(2 * 16 + l) * K + k0 + hk]);
    v16h bf3 = load_frag(&ldsB[(3 * 16 + l) * K + k0 + hk]);
    v16h afn = af;
    if (k0 + 32 < K) afn = load_frag(aRow + k0 + 32);
    acc[0] = wmma_f16(af, bf0, acc[0]);
    acc[1] = wmma_f16(af, bf1, acc[1]);
    acc[2] = wmma_f16(af, bf2, acc[2]);
    acc[3] = wmma_f16(af, bf3, acc[3]);
    af = afn;
  }
#pragma unroll
  for (int j = 0; j < 4; ++j) {
#pragma unroll
    for (int r = 0; r < 8; ++r) {
      const int rr = r + ((lane >> 4) << 3);
      Y[(size_t)(row0 + rr) * dout + col0 + j * 16 + (lane & 15)] = acc[j][r];
    }
  }
}

// --------------------------------------------------------- BN over 128k rows
__global__ void colstats(const float* __restrict__ Y, int dout,
                         float* __restrict__ sum, float* __restrict__ sq) {
  const int t    = threadIdx.x;
  const int col  = t % dout;
  const int part = t / dout;
  const int nsub = 256 / dout;         // 1 (dout=256) or 2 (dout=128)
  const int rsub = NPTS / nsub;        // 500 or 250
  const size_t base = (size_t)blockIdx.x * NPTS + (size_t)part * rsub;
  float s = 0.f, q = 0.f;
  for (int r = 0; r < rsub; ++r) {
    const float v = Y[(base + r) * dout + col];
    s += v; q += v * v;
  }
  atomicAdd(&sum[col], s);
  atomicAdd(&sq[col], q);
}

__global__ void bn_finalize(const float* __restrict__ sum, const float* __restrict__ sq,
                            const float* __restrict__ g, const float* __restrict__ beta,
                            float* __restrict__ scale, float* __restrict__ shift, int dout) {
  const int o = threadIdx.x;
  if (o >= dout) return;
  const float invM = 1.f / (float)ROWS;
  const float mean = sum[o] * invM;
  const float var  = sq[o] * invM - mean * mean;
  const float sc   = rsqrtf(var + EPSF) * g[o];
  scale[o] = sc;
  shift[o] = beta[o] - mean * sc;
}

__global__ void bnrelu_f16(const float* __restrict__ Y, const float* __restrict__ scale,
                           const float* __restrict__ shift, int dout,
                           _Float16* __restrict__ Anext) {
  const size_t N = (size_t)ROWS * dout;
  const size_t idx = (size_t)blockIdx.x * 256 + threadIdx.x;
  if (idx >= N) return;
  const int col = (int)(idx % dout);
  const float h = fmaxf(0.f, Y[idx] * scale[col] + shift[col]);
  Anext[idx] = (_Float16)h;
}

// Last soap layer: write local_feature (f32, d_out) + channel-major f16 copy
// (stride 512, zero-padded j=500..511) for the mask GEMM B operand.
__global__ void bnrelu_last(const float* __restrict__ Y, const float* __restrict__ scale,
                            const float* __restrict__ shift, float* __restrict__ lf,
                            _Float16* __restrict__ DT) {
  const size_t idx = (size_t)blockIdx.x * 256 + threadIdx.x;
  if (idx >= (size_t)Bsz * 128 * 512) return;
  const int    j = (int)(idx & 511);
  const int    c = (int)((idx >> 9) & 127);
  const size_t b = idx >> 16;
  float h = 0.f;
  if (j < NPTS) {
    const float y = Y[((size_t)b * NPTS + j) * 128 + c];
    h = fmaxf(0.f, y * scale[c] + shift[c]);
    lf[((size_t)b * 128 + c) * NPTS + j] = h;
  }
  DT[idx] = (_Float16)h;
}

// ------------------------------------------------- correlation via mask WMMA
// corr[b,c] = (1/cnt) * sum_i (sum_j m[i,j] data[j,c]) * data[i,c]
__global__ __launch_bounds__(256) void corr_kernel(
    const float* __restrict__ xyzP,
    const float* __restrict__ sqB,
    const _Float16* __restrict__ DT,   // (B,128,512) f16
    const float* __restrict__ lf,      // (B,128,500) f32
    float* __restrict__ corrOut,       // (B,128)
    float* __restrict__ Cbuf) {        // (B,256), cols 0..127
  const int b = blockIdx.x, tid = threadIdx.x;
  __shared__ float xs[512], ys[512], zs[512], sqs[512];
  __shared__ __align__(16) _Float16 mt[16 * 32];
  __shared__ float corr_s[128];
  __shared__ float cnt_s;

  for (int i = tid; i < 512; i += 256) {
    if (i < NPTS) {
      const float* p = xyzP + ((size_t)b * NPTS + i) * 3;
      xs[i] = p[0]; ys[i] = p[1]; zs[i] = p[2];
      sqs[i] = sqB[(size_t)b * NPTS + i];
    } else { xs[i] = ys[i] = zs[i] = 0.f; sqs[i] = 1e30f; }  // sentinel => m=0
  }
  if (tid < 128) corr_s[tid] = 0.f;
  if (tid == 0) cnt_s = 0.f;
  __syncthreads();

  const int wave = tid >> 5, lane = tid & 31;
  const int c0 = wave * 16;
  const int l  = lane & 15, hk = (lane >> 4) << 3;
  const int cN = c0 + l;
  float cntLocal = 0.f;

  for (int it = 0; it < 32; ++it) {
    const int i0 = it * 16;
    v8f acc = {};
    for (int js = 0; js < 16; ++js) {
      const int j0 = js * 32;
      // Issue the global B-fragment load first: its latency hides behind the
      // VALU-heavy mask-tile build below.
      const v16h bf = load_frag(DT + ((size_t)b * 128 + cN) * 512 + j0 + hk);
      for (int e = tid; e < 512; e += 256) {          // build 16x32 mask tile
        const int ii = i0 + (e >> 5);
        const int jj = j0 + (e & 31);
        const float d2 = sqs[ii] + sqs[jj]
            - 2.f * (xs[ii] * xs[jj] + ys[ii] * ys[jj] + zs[ii] * zs[jj]);
        const float dist = sqrtf(fabsf(d2));
        const float mm = (dist > 19.f && dist < 21.f) ? 1.f : 0.f;
        mt[e] = (_Float16)mm;
        cntLocal += mm;
      }
      __syncthreads();
      const v16h af = load_frag(&mt[l * 32 + hk]);                   // LDS A
      acc = wmma_f16(af, bf, acc);
      __syncthreads();
    }
    float partial = 0.f;
#pragma unroll
    for (int r = 0; r < 8; ++r) {
      const int i = i0 + r + ((lane >> 4) << 3);
      if (i < NPTS) partial += acc[r] * lf[((size_t)b * 128 + cN) * NPTS + i];
    }
    atomicAdd(&corr_s[cN], partial);
  }
  atomicAdd(&cnt_s, cntLocal);
  __syncthreads();
  if (tid < 128) {
    const float v = corr_s[tid] / cnt_s;
    corrOut[(size_t)b * 128 + tid] = v;
    Cbuf[(size_t)b * 256 + tid] = v;
  }
}

// --------------------------------------------- tiny fused FC + BN(axis0) + ReLU
__global__ void fc_bn_small(const float* __restrict__ G, int din, int in_ld,
                            const float* __restrict__ Wm,
                            const float* __restrict__ g, const float* __restrict__ beta,
                            float* __restrict__ O, int out_ld, int out_off) {
  const int o = blockIdx.x, t = threadIdx.x;
  __shared__ float wrow[256];
  __shared__ float red[256];
  for (int k = t; k < din; k += 256) wrow[k] = Wm[(size_t)o * din + k];
  __syncthreads();
  const float* gr = G + (size_t)t * in_ld;
  float y = 0.f;
  for (int k = 0; k < din; ++k) y += gr[k] * wrow[k];
  red[t] = y; __syncthreads();
  for (int s = 128; s > 0; s >>= 1) { if (t < s) red[t] += red[t + s]; __syncthreads(); }
  const float mean = red[0] / 256.f; __syncthreads();
  red[t] = y * y; __syncthreads();
  for (int s = 128; s > 0; s >>= 1) { if (t < s) red[t] += red[t + s]; __syncthreads(); }
  const float var = red[0] / 256.f - mean * mean;
  const float h = fmaxf(0.f, (y - mean) * rsqrtf(var + EPSF) * g[o] + beta[o]);
  O[(size_t)t * out_ld + out_off + o] = h;
}

__global__ void lt4_kernel(const float* __restrict__ C, const float* __restrict__ W4,
                           const float* __restrict__ b4, float* __restrict__ out) {
  const int b = blockIdx.x, t = threadIdx.x;
  __shared__ float red[256];
  red[t] = C[(size_t)b * 256 + t] * W4[t];
  __syncthreads();
  for (int s = 128; s > 0; s >>= 1) { if (t < s) red[t] += red[t + s]; __syncthreads(); }
  if (t == 0) out[b] = red[0] + b4[0];
}

// ---------------------------------------------------------------------------
extern "C" void kernel_launch(void* const* d_in, const int* in_sizes, int n_in,
                              void* d_out, int out_size, void* d_ws, size_t ws_size,
                              hipStream_t stream) {
  (void)in_sizes; (void)n_in; (void)out_size; (void)ws_size;
  const float* x = (const float*)d_in[0];
  // Param layout: soap[5]x(W,b,g,beta), mbtr[5]x4, res[3]x4, lt1 x4, lt4 (W,b)
  const float *soapW[5], *soapG[5], *soapBt[5];
  const float *mbtrW[5], *mbtrG[5], *mbtrBt[5];
  const float *resW[4],  *resG[4],  *resBt[4];   // res[0..2] + lt1 at [3]
  for (int i = 0; i < 5; ++i) {
    soapW[i] = (const float*)d_in[1 + 4 * i];
    soapG[i] = (const float*)d_in[3 + 4 * i];
    soapBt[i] = (const float*)d_in[4 + 4 * i];
    mbtrW[i] = (const float*)d_in[21 + 4 * i];
    mbtrG[i] = (const float*)d_in[23 + 4 * i];
    mbtrBt[i] = (const float*)d_in[24 + 4 * i];
  }
  for (int i = 0; i < 4; ++i) {       // 41..52 res, 53..56 lt1
    resW[i] = (const float*)d_in[41 + 4 * i];
    resG[i] = (const float*)d_in[43 + 4 * i];
    resBt[i] = (const float*)d_in[44 + 4 * i];
  }
  const float* W4 = (const float*)d_in[57];
  const float* b4 = (const float*)d_in[58];

  float* out = (float*)d_out;
  const size_t OUT_LF   = 512;
  const size_t OUT_CORR = 512 + (size_t)Bsz * 128 * NPTS;

  // Workspace carve-up (256B aligned)
  char* base = (char*)d_ws;
  size_t off = 0;
  auto alloc = [&](size_t bytes) -> char* {
    char* p = base + off;
    off = (off + bytes + 255) & ~(size_t)255;
    return p;
  };
  _Float16* Aping = (_Float16*)alloc((size_t)ROWS * 256 * 2);
  _Float16* Apong = (_Float16*)alloc((size_t)ROWS * 256 * 2);
  float*    Yb    = (float*)   alloc((size_t)ROWS * 256 * 4);
  _Float16* Wf16  = (_Float16*)alloc((size_t)256 * 256 * 2);
  float*    stats = (float*)   alloc(1024 * 4);  // sum|sq|scale|shift
  float*    xyzP  = (float*)   alloc((size_t)ROWS * 3 * 4);
  float*    sqB   = (float*)   alloc((size_t)ROWS * 4);
  _Float16* DT    = (_Float16*)alloc((size_t)Bsz * 128 * 512 * 2);
  float*    G0    = (float*)   alloc((size_t)Bsz * 157 * 4);
  float*    G1    = (float*)   alloc((size_t)Bsz * 256 * 4);
  float*    G2    = (float*)   alloc((size_t)Bsz * 256 * 4);
  float*    Cbuf  = (float*)   alloc((size_t)Bsz * 256 * 4);
  float*    Dbuf  = (float*)   alloc((size_t)Bsz * 256 * 4);
  float* statSum = stats, *statSq = stats + 256, *statSc = stats + 512, *statSh = stats + 768;

  // ---- prep
  tsys_kernel<<<Bsz, 256, 0, stream>>>(x, out);
  soap_cvt<<<(unsigned)(((size_t)ROWS * 160 + 255) / 256), 256, 0, stream>>>(x, Aping);
  xyz_sq<<<(ROWS + 255) / 256, 256, 0, stream>>>(x, xyzP, sqB);
  mbtr_cvt<<<(Bsz * 157 + 255) / 256, 256, 0, stream>>>(x, G0);

  // ---- soap MLP: 5 x (wcvt -> WMMA GEMM -> stats -> finalize -> BN/ReLU)
  const int Kpad[5] = {160, 256, 256, 256, 256};
  const int dinL[5] = {147, 256, 256, 256, 256};
  const int doutL[5] = {256, 256, 256, 256, 128};
  _Float16* bufA[2] = {Aping, Apong};
  for (int lyr = 0; lyr < 5; ++lyr) {
    const int K = Kpad[lyr], dN = doutL[lyr];
    const int colChunks = dN / 64;
    wcvt_f16<<<(dN * K + 255) / 256, 256, 0, stream>>>(soapW[lyr], dinL[lyr], K, dN, Wf16);
    dim3 ggrid(ROWS / 128, colChunks);
    gemm_wmma_f16<<<ggrid, 256, 0, stream>>>(bufA[lyr & 1], K, Wf16, K, Yb, dN, K);
    zero_f32<<<2, 256, 0, stream>>>(stats, 512);
    colstats<<<Bsz, 256, 0, stream>>>(Yb, dN, statSum, statSq);
    bn_finalize<<<1, 256, 0, stream>>>(statSum, statSq, soapG[lyr], soapBt[lyr],
                                       statSc, statSh, dN);
    if (lyr < 4) {
      bnrelu_f16<<<(unsigned)(((size_t)ROWS * dN + 255) / 256), 256, 0, stream>>>(
          Yb, statSc, statSh, dN, bufA[(lyr + 1) & 1]);
    } else {
      bnrelu_last<<<(unsigned)(((size_t)Bsz * 128 * 512) / 256), 256, 0, stream>>>(
          Yb, statSc, statSh, out + OUT_LF, DT);
    }
  }

  // ---- correlation (mask WMMA), writes d_out corr + Cbuf[:,0:128]
  corr_kernel<<<Bsz, 256, 0, stream>>>(xyzP, sqB, DT, out + OUT_LF,
                                       out + OUT_CORR, Cbuf);

  // ---- mbtr chain (256 rows), last layer into Cbuf[:,128:256]
  fc_bn_small<<<256, 256, 0, stream>>>(G0, 157, 157, mbtrW[0], mbtrG[0], mbtrBt[0], G1, 256, 0);
  fc_bn_small<<<256, 256, 0, stream>>>(G1, 256, 256, mbtrW[1], mbtrG[1], mbtrBt[1], G2, 256, 0);
  fc_bn_small<<<256, 256, 0, stream>>>(G2, 256, 256, mbtrW[2], mbtrG[2], mbtrBt[2], G1, 256, 0);
  fc_bn_small<<<256, 256, 0, stream>>>(G1, 256, 256, mbtrW[3], mbtrG[3], mbtrBt[3], G2, 256, 0);
  fc_bn_small<<<128, 256, 0, stream>>>(G2, 256, 256, mbtrW[4], mbtrG[4], mbtrBt[4], Cbuf, 256, 128);

  // ---- res x3 + lt1 + lt4
  fc_bn_small<<<256, 256, 0, stream>>>(Cbuf, 256, 256, resW[0], resG[0], resBt[0], Dbuf, 256, 0);
  fc_bn_small<<<256, 256, 0, stream>>>(Dbuf, 256, 256, resW[1], resG[1], resBt[1], Cbuf, 256, 0);
  fc_bn_small<<<256, 256, 0, stream>>>(Cbuf, 256, 256, resW[2], resG[2], resBt[2], Dbuf, 256, 0);
  fc_bn_small<<<256, 256, 0, stream>>>(Dbuf, 256, 256, resW[3], resG[3], resBt[3], Cbuf, 256, 0);
  lt4_kernel<<<Bsz, 256, 0, stream>>>(Cbuf, W4, b4, out);
}
